// MyDecoderLayerDAEFormer_1425929142550
// MI455X (gfx1250) — matrix operand order repaired and checked by reading, not compiled, not run on hardware
//
#include <hip/hip_runtime.h>

#define DIMK 256      // K (input dim)
#define NOUT 512      // N (expand dim)
#define CGRP 128      // layernorm group size
#define LN_EPS 1e-5f

typedef __attribute__((ext_vector_type(16))) __bf16 v16bf;
typedef __attribute__((ext_vector_type(8)))  float  v8f;

union AFrag {
    v16bf   v;
    unsigned u[8];
};
union BFrag {
    v16bf v;
    uint4 q[2];
};

// ---------------------------------------------------------------------------
// Prep: convert W [512,256] fp32 -> bf16, pre-swizzled into exact B-fragment
// order: Wsw[g][t][k][lane] = 16 contiguous bf16 (32 B) that lane needs for
// wmma step (t,k) of group g. One thread per (g,t,k,lane) = 8192 threads.
// ---------------------------------------------------------------------------
__global__ __launch_bounds__(256)
void prep_w_kernel(const float* __restrict__ Wx, uint4* __restrict__ Wsw)
{
    int idx  = blockIdx.x * 256 + threadIdx.x;   // 0..8191
    int lane = idx & 31;
    int k    = (idx >> 5) & 7;
    int t    = (idx >> 8) & 7;
    int g    = (idx >> 11) & 3;

    int col = g * CGRP + t * 16 + (lane & 15);        // W row (= B column)
    int kk  = k * 32 + ((lane & 16) ? 16 : 0);        // K base for this lane

    const float4* src = (const float4*)(Wx + (size_t)col * DIMK + kk);
    float4 q0 = src[0], q1 = src[1], q2 = src[2], q3 = src[3];

    BFrag bf;
    bf.v[0]  = (__bf16)q0.x; bf.v[1]  = (__bf16)q0.y;
    bf.v[2]  = (__bf16)q0.z; bf.v[3]  = (__bf16)q0.w;
    bf.v[4]  = (__bf16)q1.x; bf.v[5]  = (__bf16)q1.y;
    bf.v[6]  = (__bf16)q1.z; bf.v[7]  = (__bf16)q1.w;
    bf.v[8]  = (__bf16)q2.x; bf.v[9]  = (__bf16)q2.y;
    bf.v[10] = (__bf16)q2.z; bf.v[11] = (__bf16)q2.w;
    bf.v[12] = (__bf16)q3.x; bf.v[13] = (__bf16)q3.y;
    bf.v[14] = (__bf16)q3.z; bf.v[15] = (__bf16)q3.w;

    Wsw[(size_t)idx * 2 + 0] = bf.q[0];
    Wsw[(size_t)idx * 2 + 1] = bf.q[1];
}

// ---------------------------------------------------------------------------
// Main: block = 32 rows x 512 cols. Wave = 16 rows x one 128-ch LN group.
// ---------------------------------------------------------------------------
__global__ __launch_bounds__(256)
void patch_expand_ln_kernel(const float* __restrict__ x1,
                            const uint4* __restrict__ Wsw,
                            const float* __restrict__ gamma,
                            const float* __restrict__ beta,
                            float* __restrict__ out)
{
    __shared__ unsigned short As[32][DIMK];   // 16 KB bf16 A tile

    const int tid  = threadIdx.x;
    const int lane = tid & 31;
    const int wid  = tid >> 5;        // 0..7
    const int m    = wid >> 2;        // 0..1  : 16-row subtile
    const int g    = wid & 3;         // 0..3  : 128-col LayerNorm group
    const int r0   = blockIdx.x * 32; // first global row of this block

    // ---- Stage A tile (32 rows x 256 fp32) into LDS as bf16 ----
    {
        const float4* src = (const float4*)(x1 + (size_t)r0 * DIMK);
        #pragma unroll
        for (int j = 0; j < 8; ++j) {
            int f    = tid + j * 256;     // float4 index in tile [0,2048)
            int row  = f >> 6;            // 64 float4 per row
            int col4 = f & 63;
            float4 v = src[f];
            unsigned short b0 = __builtin_bit_cast(unsigned short, (__bf16)v.x);
            unsigned short b1 = __builtin_bit_cast(unsigned short, (__bf16)v.y);
            unsigned short b2 = __builtin_bit_cast(unsigned short, (__bf16)v.z);
            unsigned short b3 = __builtin_bit_cast(unsigned short, (__bf16)v.w);
            unsigned* dst = (unsigned*)&As[row][col4 * 4];
            dst[0] = (unsigned)b0 | ((unsigned)b1 << 16);
            dst[1] = (unsigned)b2 | ((unsigned)b3 << 16);
        }
    }

    // Per-wave B base: Wsw + (g*64*32 + lane) fragments (2 uint4 each).
    const uint4* bbase = Wsw + ((size_t)(g * 2048 + lane)) * 2;
    // Warm the cache for this wave's swizzled-W region (global_prefetch_b8)
    __builtin_prefetch(bbase, 0, 1);

    __syncthreads();

    // ---- Load all 8 A fragments (K=0..255) for this wave from LDS ----
    // k order used here (same as the B packing): lanes 0-15 hold K=kb..kb+15 contiguous (VGPR0-7);
    // lanes 16-31 hold K=kb+16..kb+31 -> two aligned b128 LDS reads.
    const int arow = m * 16 + (lane & 15);
    AFrag afrag[8];
    #pragma unroll
    for (int k = 0; k < 8; ++k) {
        int kb = k * 32 + ((lane & 16) ? 16 : 0);
        uint4 c0 = *(const uint4*)&As[arow][kb];
        uint4 c1 = *(const uint4*)&As[arow][kb + 8];
        afrag[k].u[0] = c0.x; afrag[k].u[1] = c0.y;
        afrag[k].u[2] = c0.z; afrag[k].u[3] = c0.w;
        afrag[k].u[4] = c1.x; afrag[k].u[5] = c1.y;
        afrag[k].u[6] = c1.z; afrag[k].u[7] = c1.w;
    }

    // ---- GEMM: this wave computes 16 rows x 128 cols, K = 256 ----
    v8f zero = {};
    v8f acc[8];
    #pragma unroll
    for (int t = 0; t < 8; ++t) acc[t] = zero;

    #pragma unroll
    for (int t = 0; t < 8; ++t) {
        #pragma unroll
        for (int k = 0; k < 8; ++k) {
            // B fragment: two b128 loads at compile-time immediate offsets,
            // coalesced 1 KB per (t,k) across the wave.
            BFrag bf;
            bf.q[0] = bbase[(t * 8 + k) * 64 + 0];
            bf.q[1] = bbase[(t * 8 + k) * 64 + 1];
            acc[t] = __builtin_amdgcn_wmma_f32_16x16x32_bf16(
                false, afrag[k].v, false, bf.v, (short)0, acc[t], false, false);
        }
    }

    // ---- LayerNorm over the 128 channels of this wave's group ----
    // Row M lives in VGPR r = M%8, lane half = M/8; reduce across 16 lanes.
    float mu[8], rs[8];
    #pragma unroll
    for (int r = 0; r < 8; ++r) {
        float s = 0.f, q = 0.f;
        #pragma unroll
        for (int t = 0; t < 8; ++t) { float v = acc[t][r]; s += v; q += v * v; }
        #pragma unroll
        for (int mask = 1; mask < 16; mask <<= 1) {
            s += __shfl_xor(s, mask, 32);
            q += __shfl_xor(q, mask, 32);
        }
        float mean = s * (1.0f / 128.0f);
        float var  = q * (1.0f / 128.0f) - mean * mean;
        mu[r] = mean;
        rs[r] = rsqrtf(var + LN_EPS);
    }

    // gamma/beta for this lane's channels (cc = t*16 + lane%16)
    float gv[8], bv[8];
    #pragma unroll
    for (int t = 0; t < 8; ++t) {
        int cc = t * 16 + (lane & 15);
        gv[t] = gamma[cc];
        bv[t] = beta[cc];
    }

    // ---- Pixel-shuffle scatter: out[b, (2h+p1)*128 + (2w+p2), cc] ----
    const int p1 = g >> 1, p2 = g & 1;
    #pragma unroll
    for (int r = 0; r < 8; ++r) {
        int M   = r + ((lane & 16) ? 8 : 0);
        int l   = r0 + m * 16 + M;       // global input row
        int b   = l >> 12;               // /4096
        int pos = l & 4095;
        int h   = pos >> 6, w = pos & 63;
        int row2 = (2 * h + p1) * 128 + (2 * w + p2);
        float* orow = out + ((size_t)b * 16384 + (size_t)row2) * 128 + (lane & 15);
        #pragma unroll
        for (int t = 0; t < 8; ++t) {
            orow[t * 16] = (acc[t][r] - mu[r]) * rs[r] * gv[t] + bv[t];
        }
    }
}

extern "C" void kernel_launch(void* const* d_in, const int* in_sizes, int n_in,
                              void* d_out, int out_size, void* d_ws, size_t ws_size,
                              hipStream_t stream) {
    const float* x1    = (const float*)d_in[0];
    const float* Wx    = (const float*)d_in[1];
    const float* gamma = (const float*)d_in[2];
    const float* beta  = (const float*)d_in[3];
    float* out  = (float*)d_out;
    uint4* Wsw  = (uint4*)d_ws;    // 256 KB swizzled bf16 W

    // 1) one-shot W convert+swizzle (runs every call for determinism)
    prep_w_kernel<<<dim3(32), dim3(256), 0, stream>>>(Wx, Wsw);

    // 2) fused expand + pixel-shuffle + LayerNorm
    const int total_rows = in_sizes[0] / DIMK;   // 16*4096 = 65536
    patch_expand_ln_kernel<<<dim3(total_rows / 32), dim3(256), 0, stream>>>(
        x1, Wsw, gamma, beta, out);
}
